// MultiHeadCrossAttention_50319836840451
// MI455X (gfx1250) — matrix operand order repaired and checked by reading, not compiled, not run on hardware
//
#include <hip/hip_runtime.h>
#include <hip/hip_bf16.h>
#include <stdint.h>

// ---------------------------------------------------------------------------
// MI455X (gfx1250) multi-head attention, all matmuls via v_wmma_f32_16x16x32_bf16
// B=2, S=2048, D=1024, H=16, HD=64
// Attention computes S^T = K@Q^T and ctx^T = V^T@P^T so softmax reductions are
// per-lane; 32-query tiles reuse each K/V fragment for two WMMAs.
// ---------------------------------------------------------------------------

typedef __bf16 bf16_t;
typedef __attribute__((ext_vector_type(16))) __bf16 v16bf;
typedef __attribute__((ext_vector_type(8)))  __bf16 bf16x8;
typedef __attribute__((ext_vector_type(8)))  float  v8f;

constexpr int Bc  = 2;
constexpr int Sc  = 2048;
constexpr int Dc  = 1024;
constexpr int Hc  = 16;
constexpr int HDc = 64;
constexpr int Mrows = Bc * Sc;      // 4096 token rows

// A-fragment (16x32 bf16, row-major source, K contiguous):
// lane l: m = l&15, hi = l>>4 ; element e: K = kbase + e + 16*hi
__device__ inline v16bf load_a_frag(const bf16_t* __restrict__ rowptr, int kbase, int hi) {
    bf16x8 lo = *reinterpret_cast<const bf16x8*>(rowptr + kbase + hi * 16);
    bf16x8 h8 = *reinterpret_cast<const bf16x8*>(rowptr + kbase + hi * 16 + 8);
    v16bf a;
#pragma unroll
    for (int i = 0; i < 8; ++i) { a[i] = lo[i]; a[8 + i] = h8[i]; }
    return a;
}

__device__ inline v8f zero_v8f() {
    v8f z;
#pragma unroll
    for (int i = 0; i < 8; ++i) z[i] = 0.0f;
    return z;
}

// round-to-nearest-even fp32->bf16, packed pair into one dword
__device__ inline uint32_t pack_bf16(float a, float b) {
    union { float f; uint32_t u; } ua, ub;
    ua.f = a; ub.f = b;
    uint32_t ha = (ua.u + 0x7FFFu + ((ua.u >> 16) & 1u)) >> 16;
    uint32_t hb = (ub.u + 0x7FFFu + ((ub.u >> 16) & 1u)) >> 16;
    return ha | (hb << 16);
}

// ---------------------------------------------------------------------------
// Elementwise fp32 -> bf16
// ---------------------------------------------------------------------------
__global__ void __launch_bounds__(256) k_to_bf16(const float* __restrict__ src,
                                                 bf16_t* __restrict__ dst, int count) {
    int i = blockIdx.x * 256 + threadIdx.x;
    if (i < count) dst[i] = (bf16_t)src[i];
}

// WT[n*Kd + k] = (bf16) W[k*N + n]   (transpose so B-fragments load contiguously)
__global__ void __launch_bounds__(256) k_transpose_bf16(const float* __restrict__ W,
                                                        bf16_t* __restrict__ WT,
                                                        int Kd, int N) {
    int i = blockIdx.x * 256 + threadIdx.x;
    if (i >= Kd * N) return;
    int k = i / N, n = i % N;
    WT[(size_t)n * Kd + k] = (bf16_t)W[i];
}

// ---------------------------------------------------------------------------
// Generic WMMA GEMM: C[M,N] = A[M,Kd] @ WT^T + bias
//   A: bf16 row-major [M,Kd]; WT: bf16 [N,Kd] (transposed weight)
//   One wave computes a 32x64 output tile (2x4 accum tiles): each B-fragment
//   is reused by two WMMAs -> 8 wmma per (2 A-frag + 4 B-frag) loads.
// Epilogue MODE: 0 = fp32 row-major out
//                1 = Q  -> bf16 [B,H,S,HD]
//                2 = KV -> K bf16 [B,H,S,HD], V bf16 transposed [B,H,HD,S]
// ---------------------------------------------------------------------------
template <int MODE>
__global__ void __launch_bounds__(256)
k_gemm_wmma(const bf16_t* __restrict__ A, const bf16_t* __restrict__ WT,
            const float* __restrict__ bias, int Kd, int Ntiles,
            float* __restrict__ outF, bf16_t* __restrict__ outQ,
            bf16_t* __restrict__ outK, bf16_t* __restrict__ outVT) {
    const int wid  = (blockIdx.x * 256 + (int)threadIdx.x) >> 5;
    const int lane = threadIdx.x & 31;
    const int n    = lane & 15;
    const int hi   = lane >> 4;
    const int nt   = wid % Ntiles;   // 64-wide column tile
    const int mt   = wid / Ntiles;   // 32-high row tile

    const bf16_t* arow0 = A + (size_t)(mt * 32 + n) * Kd;        // rows 0..15
    const bf16_t* arow1 = A + (size_t)(mt * 32 + 16 + n) * Kd;   // rows 16..31

    v8f acc[2][4];
#pragma unroll
    for (int g = 0; g < 2; ++g)
#pragma unroll
        for (int t = 0; t < 4; ++t) acc[g][t] = zero_v8f();

    for (int kb = 0; kb < Kd; kb += 32) {
        v16bf af0 = load_a_frag(arow0, kb, hi);
        v16bf af1 = load_a_frag(arow1, kb, hi);
#pragma unroll
        for (int t = 0; t < 4; ++t) {
            const bf16_t* bp = WT + (size_t)(nt * 64 + t * 16 + n) * Kd + kb + hi * 16;
            v16bf bf_ = *reinterpret_cast<const v16bf*>(bp);
            acc[0][t] = __builtin_amdgcn_wmma_f32_16x16x32_bf16(
                false, af0, false, bf_, (short)0, acc[0][t], false, false);
            acc[1][t] = __builtin_amdgcn_wmma_f32_16x16x32_bf16(
                false, af1, false, bf_, (short)0, acc[1][t], false, false);
        }
    }

#pragma unroll
    for (int g = 0; g < 2; ++g)
#pragma unroll
    for (int t = 0; t < 4; ++t) {
        const int col = nt * 64 + t * 16 + n;
        const float bv = bias[col];
#pragma unroll
        for (int r = 0; r < 8; ++r) {
            const int row = mt * 32 + g * 16 + r + hi * 8;   // C/D: M = r + 8*hi
            const float v = acc[g][t][r] + bv;
            if (MODE == 0) {
                outF[(size_t)row * (Ntiles * 64) + col] = v;
            } else if (MODE == 1) {
                const int b = row / Sc, s = row % Sc;
                const int h = col / HDc, d = col % HDc;
                outQ[((size_t)(b * Hc + h) * Sc + s) * HDc + d] = (bf16_t)v;
            } else {
                const int b = row / Sc, s = row % Sc;
                const int h = col / (2 * HDc), rr = col % (2 * HDc);
                if (rr < HDc)
                    outK[((size_t)(b * Hc + h) * Sc + s) * HDc + rr] = (bf16_t)v;
                else
                    outVT[((size_t)(b * Hc + h) * HDc + (rr - HDc)) * Sc + s] = (bf16_t)v;
            }
        }
    }
}

// ---------------------------------------------------------------------------
// Online-softmax step for one 16-query group over a 32-key block.
// st0/st1: S^T accum tiles (lane = query, reg r -> key r+8*hi [+16 for st1]).
// Updates running max/sum and rescales the 4 ctx^T accum tiles; returns the
// P^T B-fragment (lane = query, elem e <-> key = e + 16*hi).
// ---------------------------------------------------------------------------
__device__ inline v16bf softmax_update(const v8f& st0, const v8f& st1,
                                       float& m_, float& l_, v8f acc[4], int hi) {
    const float scale = 0.125f;  // 1/sqrt(64)
    float s0[8], s1[8];
    float mx = -1.0e30f;
#pragma unroll
    for (int r = 0; r < 8; ++r) {
        s0[r] = st0[r] * scale;
        s1[r] = st1[r] * scale;
        mx = fmaxf(mx, fmaxf(s0[r], s1[r]));
    }
    mx = fmaxf(mx, __shfl_xor(mx, 16, 32));
    const float nm = fmaxf(m_, mx);
    const float alpha = __expf(m_ - nm);
    m_ = nm;

    float p0[8], p1[8], rs = 0.0f;
#pragma unroll
    for (int r = 0; r < 8; ++r) {
        p0[r] = __expf(s0[r] - nm);
        p1[r] = __expf(s1[r] - nm);
        rs += p0[r] + p1[r];
    }
    rs += __shfl_xor(rs, 16, 32);
    l_ = l_ * alpha + rs;

#pragma unroll
    for (int t = 0; t < 4; ++t)
#pragma unroll
        for (int r = 0; r < 8; ++r) acc[t][r] *= alpha;

    uint32_t own0[4], own1[4], oth0[4], oth1[4];
#pragma unroll
    for (int r = 0; r < 4; ++r) {
        own0[r] = pack_bf16(p0[2 * r], p0[2 * r + 1]);
        own1[r] = pack_bf16(p1[2 * r], p1[2 * r + 1]);
    }
#pragma unroll
    for (int r = 0; r < 4; ++r) {
        oth0[r] = __shfl_xor(own0[r], 16, 32);
        oth1[r] = __shfl_xor(own1[r], 16, 32);
    }
    union { uint32_t u[8]; v16bf v; } pf;
#pragma unroll
    for (int r = 0; r < 4; ++r) {
        // hi==0: elems 0..7 = keys 0..7   (own t0), elems 8..15 = keys 8..15  (other t0)
        // hi==1: elems 0..7 = keys 16..23 (other t1), elems 8..15 = keys 24..31 (own t1)
        pf.u[r]     = hi ? oth1[r] : own0[r];
        pf.u[4 + r] = hi ? own1[r] : oth0[r];
    }
    return pf.v;
}

// ---------------------------------------------------------------------------
// Flash attention (transposed tiles): one wave per (b,h, 32-query block).
// Per 32-key block: 8 wmma for S^T (K fragments shared by 2 query groups),
// per-group softmax, 8 wmma for ctx^T (V fragments shared by 2 groups).
// ---------------------------------------------------------------------------
__global__ void __launch_bounds__(256)
k_attn_wmma(const bf16_t* __restrict__ Q, const bf16_t* __restrict__ K,
            const bf16_t* __restrict__ VT, bf16_t* __restrict__ ctx) {
    const int widL = threadIdx.x >> 5;
    const int wid  = blockIdx.x * 8 + widL;
    const int lane = threadIdx.x & 31;
    const int n    = lane & 15;      // query within group / d within tile
    const int hi   = lane >> 4;
    const int bh   = wid >> 6;       // / (S/32)
    const int qb   = wid & 63;

    const bf16_t* Qp = Q  + ((size_t)bh * Sc + qb * 32) * HDc;
    const bf16_t* Kp = K  + (size_t)bh * Sc * HDc;
    const bf16_t* Vp = VT + (size_t)bh * HDc * Sc;

    // Q^T B-fragments for 2 query groups x 2 K-chunks (lane n = query)
    v16bf qf[2][2];
#pragma unroll
    for (int g = 0; g < 2; ++g) {
        const bf16_t* qr = Qp + (size_t)(g * 16 + n) * HDc;
        qf[g][0] = *reinterpret_cast<const v16bf*>(qr + 0  + hi * 16);
        qf[g][1] = *reinterpret_cast<const v16bf*>(qr + 32 + hi * 16);
    }

    float m_[2] = {-1.0e30f, -1.0e30f}, l_[2] = {0.0f, 0.0f};
    v8f acc[2][4];                   // [query group][d tile]
#pragma unroll
    for (int g = 0; g < 2; ++g)
#pragma unroll
        for (int t = 0; t < 4; ++t) acc[g][t] = zero_v8f();

    for (int kb = 0; kb < Sc; kb += 32) {
        // ---- K A-fragments: 2 key tiles x 2 chunks, reused by both groups ----
        v16bf ka[2][2];
#pragma unroll
        for (int kt = 0; kt < 2; ++kt) {
            const bf16_t* kr = Kp + (size_t)(kb + kt * 16 + n) * HDc;
            ka[kt][0] = load_a_frag(kr, 0,  hi);
            ka[kt][1] = load_a_frag(kr, 32, hi);
        }
        // ---- S^T = K @ Q^T : 8 wmma ----
        v8f st[2][2];
#pragma unroll
        for (int g = 0; g < 2; ++g)
#pragma unroll
            for (int kt = 0; kt < 2; ++kt) {
                v8f s = zero_v8f();
                s = __builtin_amdgcn_wmma_f32_16x16x32_bf16(
                    false, ka[kt][0], false, qf[g][0], (short)0, s, false, false);
                s = __builtin_amdgcn_wmma_f32_16x16x32_bf16(
                    false, ka[kt][1], false, qf[g][1], (short)0, s, false, false);
                st[g][kt] = s;
            }

        // ---- softmax + P^T fragments per group ----
        v16bf pfv[2];
#pragma unroll
        for (int g = 0; g < 2; ++g)
            pfv[g] = softmax_update(st[g][0], st[g][1], m_[g], l_[g], acc[g], hi);

        // ---- ctx^T += V^T @ P^T : V fragments reused by both groups ----
#pragma unroll
        for (int t = 0; t < 4; ++t) {
            v16bf vf = load_a_frag(Vp + (size_t)(t * 16 + n) * Sc, kb, hi);
#pragma unroll
            for (int g = 0; g < 2; ++g)
                acc[g][t] = __builtin_amdgcn_wmma_f32_16x16x32_bf16(
                    false, vf, false, pfv[g], (short)0, acc[g][t], false, false);
        }
    }

    // ---- finalize: ctx^T / l ; d is contiguous per lane -> packed b128 stores ----
    const int b = bh >> 4, h = bh & 15;
#pragma unroll
    for (int g = 0; g < 2; ++g) {
        const float rinv = 1.0f / l_[g];
        const int q = qb * 32 + g * 16 + n;
        bf16_t* cp = ctx + ((size_t)(b * Sc + q)) * Dc + h * HDc;
#pragma unroll
        for (int t = 0; t < 4; ++t) {
            bf16x8 o;
#pragma unroll
            for (int r = 0; r < 8; ++r) o[r] = (bf16_t)(acc[g][t][r] * rinv);
            *reinterpret_cast<bf16x8*>(cp + t * 16 + hi * 8) = o;
        }
    }
}

// ---------------------------------------------------------------------------
// Launch
// ---------------------------------------------------------------------------
extern "C" void kernel_launch(void* const* d_in, const int* in_sizes, int n_in,
                              void* d_out, int out_size, void* d_ws, size_t ws_size,
                              hipStream_t stream) {
    const float* emb  = (const float*)d_in[0];
    const float* W_kv = (const float*)d_in[1];
    const float* b_kv = (const float*)d_in[2];
    const float* W_q  = (const float*)d_in[3];
    const float* b_q  = (const float*)d_in[4];
    const float* W_o  = (const float*)d_in[5];
    const float* b_o  = (const float*)d_in[6];
    float* out = (float*)d_out;

    // workspace partition (bf16 elements)
    bf16_t* p = (bf16_t*)d_ws;
    bf16_t* E16   = p; p += (size_t)Mrows * Dc;        // 4096x1024
    bf16_t* WkvT  = p; p += (size_t)(2 * Dc) * Dc;     // 2048x1024
    bf16_t* WqT   = p; p += (size_t)Dc * Dc;           // 1024x1024
    bf16_t* WoT   = p; p += (size_t)Dc * Dc;           // 1024x1024
    bf16_t* Qb    = p; p += (size_t)Bc * Hc * Sc * HDc; // [B,H,S,HD]
    bf16_t* Kb    = p; p += (size_t)Bc * Hc * Sc * HDc; // [B,H,S,HD]
    bf16_t* VTb   = p; p += (size_t)Bc * Hc * HDc * Sc; // [B,H,HD,S]
    bf16_t* ctx16 = p;                                  // 4096x1024

    // stage 0: conversions
    {
        int cnt = Mrows * Dc;
        k_to_bf16<<<(cnt + 255) / 256, 256, 0, stream>>>(emb, E16, cnt);
        int tkv = Dc * 2 * Dc;
        k_transpose_bf16<<<(tkv + 255) / 256, 256, 0, stream>>>(W_kv, WkvT, Dc, 2 * Dc);
        int tq = Dc * Dc;
        k_transpose_bf16<<<(tq + 255) / 256, 256, 0, stream>>>(W_q, WqT, Dc, Dc);
        k_transpose_bf16<<<(tq + 255) / 256, 256, 0, stream>>>(W_o, WoT, Dc, Dc);
    }

    // stage 1: projections (wave = 32x64 tile, 8 waves/block)
    {
        int wavesKV = (Mrows / 32) * (2 * Dc / 64);   // 128*32 = 4096
        k_gemm_wmma<2><<<wavesKV / 8, 256, 0, stream>>>(
            E16, WkvT, b_kv, Dc, 2 * Dc / 64, nullptr, nullptr, Kb, VTb);
        int wavesQ = (Mrows / 32) * (Dc / 64);        // 128*16 = 2048
        k_gemm_wmma<1><<<wavesQ / 8, 256, 0, stream>>>(
            E16, WqT, b_q, Dc, Dc / 64, nullptr, Qb, nullptr, nullptr);
    }

    // stage 2: flash attention (wave = one 32-query block of one (b,h))
    {
        int waves = Bc * Hc * (Sc / 32);              // 32*64 = 2048
        k_attn_wmma<<<waves / 8, 256, 0, stream>>>(Qb, Kb, VTb, ctx16);
    }

    // stage 3: output projection -> fp32
    {
        int waves = (Mrows / 32) * (Dc / 64);
        k_gemm_wmma<0><<<waves / 8, 256, 0, stream>>>(
            ctx16, WoT, b_o, Dc, Dc / 64, out, nullptr, nullptr, nullptr);
    }
    (void)in_sizes; (void)n_in; (void)out_size; (void)ws_size;
}
